// NaiveWanSelfAttention_10960756540141
// MI455X (gfx1250) — compile-verified, matrix operand order checked by reading, not run on hardware
//
#include <hip/hip_runtime.h>
#include <stdint.h>

// ---------------------------------------------------------------------------
// Problem constants (B=1)
// ---------------------------------------------------------------------------
#define S_   2048
#define H_   16
#define D_   128
#define HID_ 2048
#define N3_  (3 * HID_)
#define EPS_ 1e-6f

typedef __bf16 bf16_t;
typedef __attribute__((ext_vector_type(8)))  __bf16 bf16x8;
typedef __attribute__((ext_vector_type(16))) __bf16 bf16x16;
typedef __attribute__((ext_vector_type(8)))  float  floatx8;
typedef __attribute__((ext_vector_type(4)))  int    int4v;
typedef __attribute__((ext_vector_type(8)))  int    int8v;

// ---------------------------------------------------------------------------
// WMMA helpers (CDNA5 wave32, 16x16x32 bf16 -> f32)
// ---------------------------------------------------------------------------
__device__ __forceinline__ floatx8 wmma_bf16(bf16x16 a, bf16x16 b, floatx8 c) {
    return __builtin_amdgcn_wmma_f32_16x16x32_bf16(
        /*neg_a=*/false, a, /*neg_b=*/false, b,
        /*c_mod=*/(short)0, c, /*reuse_a=*/false, /*reuse_b=*/false);
}

// Load one 16(M)x32(K) bf16 fragment from a K-contiguous tile.
// ISA layout (16-bit A 16x32): lane 0-15 -> row=lane, K = 0..7 & 16..23
//                              lane 16-31 -> row=lane-16, K = 8..15 & 24..31
__device__ __forceinline__ bf16x16 load_frag(const bf16_t* base, int ld) {
    const int lane = threadIdx.x & 31;
    const int row  = lane & 15;
    const int kb   = (lane < 16) ? 0 : 8;
    const bf16_t* p = base + row * ld + kb;
    bf16x8 lo = *(const bf16x8*)(p);
    bf16x8 hi = *(const bf16x8*)(p + 16);
    bf16x16 r;
#pragma unroll
    for (int i = 0; i < 8; ++i) { r[i] = lo[i]; r[8 + i] = hi[i]; }
    return r;
}

// ---------------------------------------------------------------------------
// Tensor Data Mover: 2-D tile DMA, global -> LDS (bf16 elements, data_size=2B)
// D# built per CDNA5 ISA 8.3/8.4; 2-group form (tensors up to 2D).
// Issue from ONE wave only; completion via s_wait_tensorcnt.
// ---------------------------------------------------------------------------
__device__ __forceinline__ void tdm_load_2d(uint32_t lds_addr, const void* gptr,
                                            uint32_t tile_d0, uint32_t tile_d1,
                                            uint32_t tensor_d0, uint32_t tensor_d1,
                                            uint64_t stride_d0 /* elements */) {
    const uint64_t ga = (uint64_t)(uintptr_t)gptr;
    int4v g0;
    g0[0] = 1;                                            // count=1, user mode
    g0[1] = (int)lds_addr;                                // lds_addr [63:32]
    g0[2] = (int)(uint32_t)ga;                            // global_addr [95:64]
    g0[3] = (int)(((uint32_t)(ga >> 32) & 0x01ffffffu)    // global_addr [120:96]
                  | 0x80000000u);                         // type=2 ("image")
    int8v g1;
    g1[0] = 0x00010000;                                   // data_size=1 -> 2 bytes
    g1[1] = (int)((tensor_d0 & 0xffffu) << 16);           // tensor_dim0[15:0]
    g1[2] = (int)(((tensor_d0 >> 16) & 0xffffu)
                  | ((tensor_d1 & 0xffffu) << 16));       // dim0[31:16] | dim1[15:0]
    g1[3] = (int)(((tensor_d1 >> 16) & 0xffffu)
                  | (tile_d0 << 16));                     // dim1[31:16] | tile_dim0
    g1[4] = (int)(tile_d1 & 0xffffu);                     // tile_dim1, tile_dim2=0
    g1[5] = (int)(uint32_t)stride_d0;                     // tensor_dim0_stride[31:0]
    g1[6] = (int)((uint32_t)(stride_d0 >> 32) & 0xffffu); // stride[47:32]
    g1[7] = 0;
    asm volatile("tensor_load_to_lds %0, %1" :: "s"(g0), "s"(g1) : "memory");
}

__device__ __forceinline__ void tdm_wait0() {
    __builtin_amdgcn_s_wait_tensorcnt(0);
}

// ---------------------------------------------------------------------------
// f32 -> bf16 streaming convert (vector x4)
// ---------------------------------------------------------------------------
__global__ __launch_bounds__(256)
void k_f32_to_bf16(const float* __restrict__ in, bf16_t* __restrict__ out, long n) {
    long i = ((long)blockIdx.x * blockDim.x + threadIdx.x) * 4;
    if (i + 3 < n) {
        float4 f = *(const float4*)(in + i);
        out[i + 0] = (bf16_t)f.x;
        out[i + 1] = (bf16_t)f.y;
        out[i + 2] = (bf16_t)f.z;
        out[i + 3] = (bf16_t)f.w;
    }
}

// ---------------------------------------------------------------------------
// Tiled bf16 GEMM:  C(f32, MxN) = A(bf16, MxK row-major) * B(bf16, KxN) + bias
// Block tile 64x128, K step 32.  8 waves (2x4), each wave 32x32 (2x2 WMMA).
// A tile staged by the Tensor Data Mover; B tile staged transposed by VALU.
// ---------------------------------------------------------------------------
__global__ __launch_bounds__(256)
void k_gemm_bf16(const bf16_t* __restrict__ A, const bf16_t* __restrict__ B,
                 const float* __restrict__ bias, float* __restrict__ C,
                 int M, int N, int K) {
    __shared__ bf16_t sA[64][32];    // [m][k]   4 KB  (TDM destination)
    __shared__ bf16_t sB[128][32];   // [n][k]   8 KB  (B staged transposed)

    const int tid  = threadIdx.x;
    const int wave = tid >> 5;
    const int lane = tid & 31;
    const int wm   = wave >> 2;      // 0..1
    const int wn   = wave & 3;       // 0..3
    const int bm   = blockIdx.y * 64;
    const int bn   = blockIdx.x * 128;

    floatx8 acc[2][2];
#pragma unroll
    for (int i = 0; i < 2; ++i)
#pragma unroll
        for (int j = 0; j < 2; ++j) acc[i][j] = {};

    for (int k0 = 0; k0 < K; k0 += 32) {
        // stage A via TDM: 2-D tile, 32 contiguous elems x 64 rows, row stride K
        if (wave == 0) {
            tdm_load_2d((uint32_t)(uintptr_t)&sA[0][0],
                        A + (long)bm * K + k0,
                        /*tile_d0=*/32, /*tile_d1=*/64,
                        /*tensor_d0=*/(uint32_t)K, /*tensor_d1=*/(uint32_t)M,
                        /*stride_d0=*/(uint64_t)K);
        }
        // stage B transposed: 32x128 elems, 16 per thread
        {
            int idx = tid * 16;
            int r = idx >> 7, c = idx & 127;
            const bf16_t* src = B + (long)(k0 + r) * N + bn + c;
            bf16x8 d0 = *(const bf16x8*)(src);
            bf16x8 d1 = *(const bf16x8*)(src + 8);
#pragma unroll
            for (int i = 0; i < 8; ++i) { sB[c + i][r] = d0[i]; sB[c + 8 + i][r] = d1[i]; }
        }
        if (wave == 0) tdm_wait0();
        __syncthreads();

        bf16x16 af[2], bf[2];
#pragma unroll
        for (int t = 0; t < 2; ++t) af[t] = load_frag(&sA[wm * 32 + t * 16][0], 32);
#pragma unroll
        for (int t = 0; t < 2; ++t) bf[t] = load_frag(&sB[wn * 32 + t * 16][0], 32);
#pragma unroll
        for (int tm = 0; tm < 2; ++tm)
#pragma unroll
            for (int tn = 0; tn < 2; ++tn)
                acc[tm][tn] = wmma_bf16(af[tm], bf[tn], acc[tm][tn]);
        __syncthreads();
    }

    // epilogue: C-layout row = r + 8*(lane>=16), col = lane&15.
    // One base pointer + increments; stores use immediate offsets.
    const int half = lane >> 4;
    const int col  = lane & 15;
    const int cn   = bn + wn * 32 + col;
    const float b0 = bias[cn];
    const float b1 = bias[cn + 16];
    float* p = C + (long)(bm + wm * 32 + 8 * half) * N + cn;
#pragma unroll
    for (int tm = 0; tm < 2; ++tm) {
#pragma unroll
        for (int r = 0; r < 8; ++r) {
            p[0]  = acc[tm][0][r] + b0;
            p[16] = acc[tm][1][r] + b1;
            p += N;
        }
        p += (long)8 * N;   // skip other half's 8 rows to next 16-row tile
    }
}

// ---------------------------------------------------------------------------
// RMSNorm(q,k) + RoPE + head-major transpose, f32 -> bf16.
// One block per sequence position; 256 threads x 8 elems = 2048.
// Outputs q/k/v as [H][S][D] bf16.
// ---------------------------------------------------------------------------
__device__ __forceinline__ float block_sum(float x, float* red) {
    int tid = threadIdx.x;
    red[tid] = x;
    __syncthreads();
    for (int s = 128; s > 0; s >>= 1) {
        if (tid < s) red[tid] += red[tid + s];
        __syncthreads();
    }
    float r = red[0];
    __syncthreads();
    return r;
}

__global__ __launch_bounds__(256)
void k_norm_rope(const float* __restrict__ qkv,
                 const float* __restrict__ cosb, const float* __restrict__ sinb,
                 const float* __restrict__ wq, const float* __restrict__ wk,
                 bf16_t* __restrict__ qo, bf16_t* __restrict__ ko,
                 bf16_t* __restrict__ vo) {
    __shared__ float red[256];
    const int s    = blockIdx.x;
    const int base = threadIdx.x * 8;

    const float* qrow = qkv + (long)s * N3_;
    const float* krow = qrow + HID_;
    const float* vrow = krow + HID_;

    float q[8], k[8];
    float sq = 0.f, sk = 0.f;
#pragma unroll
    for (int i = 0; i < 8; ++i) {
        q[i] = qrow[base + i]; sq += q[i] * q[i];
        k[i] = krow[base + i]; sk += k[i] * k[i];
    }
    float tq = block_sum(sq, red);
    float tk = block_sum(sk, red);
    float rq = rsqrtf(tq * (1.0f / HID_) + EPS_);
    float rk = rsqrtf(tk * (1.0f / HID_) + EPS_);

#pragma unroll
    for (int j = 0; j < 4; ++j) {                  // 4 rope pairs per thread
        int e  = base + 2 * j;
        int h  = e >> 7;
        int d  = e & 127;
        int jj = d >> 1;
        float c  = cosb[s * (D_ / 2) + jj];
        float sn = sinb[s * (D_ / 2) + jj];
        long off = ((long)h * S_ + s) * D_ + d;

        float x1 = q[2 * j] * rq * wq[e];
        float x2 = q[2 * j + 1] * rq * wq[e + 1];
        qo[off]     = (bf16_t)(x1 * c - x2 * sn);
        qo[off + 1] = (bf16_t)(x1 * sn + x2 * c);

        float y1 = k[2 * j] * rk * wk[e];
        float y2 = k[2 * j + 1] * rk * wk[e + 1];
        ko[off]     = (bf16_t)(y1 * c - y2 * sn);
        ko[off + 1] = (bf16_t)(y1 * sn + y2 * c);
    }
#pragma unroll
    for (int i = 0; i < 8; ++i) {
        int e = base + i;
        int h = e >> 7, d = e & 127;
        vo[((long)h * S_ + s) * D_ + d] = (bf16_t)vrow[e];
    }
}

// ---------------------------------------------------------------------------
// Flash attention (non-causal).  Grid: (S/128, H).  8 waves, 16 q-rows/wave.
// KV tile = 32.  scores: 2x4 WMMA;  P*V: 8 WMMA.  Online softmax in f32.
// K tile staged by TDM (contiguous 32x128); V tile staged transposed by VALU.
// Output written as bf16 [S][HID] for the final GEMM.
// ---------------------------------------------------------------------------
__global__ __launch_bounds__(256)
void k_attn(const bf16_t* __restrict__ Q, const bf16_t* __restrict__ Kh,
            const bf16_t* __restrict__ V, bf16_t* __restrict__ O) {
    __shared__ bf16_t sK[32][128];       // [kv][d]     8 KB  (TDM destination)
    __shared__ bf16_t sVt[128][32];      // [d][kv]     8 KB
    __shared__ bf16_t sP[8][16][32];     // per-wave P  8 KB

    const int h     = blockIdx.y;
    const int tid   = threadIdx.x;
    const int wave  = tid >> 5;
    const int lane  = tid & 31;
    const int half  = lane >> 4;
    const int col   = lane & 15;
    const int qrow0 = blockIdx.x * 128 + wave * 16;
    const float scale = 0.08838834764831845f;    // 1/sqrt(128)

    // Q fragments loaded once, straight from global (row-major [s][d])
    const bf16_t* Qp = Q + ((long)h * S_ + qrow0) * D_;
    bf16x16 qf[4];
#pragma unroll
    for (int dk = 0; dk < 4; ++dk) qf[dk] = load_frag(Qp + dk * 32, D_);

    float m[8], l[8];
    floatx8 o[8];
#pragma unroll
    for (int r = 0; r < 8; ++r) { m[r] = -3.0e38f; l[r] = 0.f; }
#pragma unroll
    for (int nt = 0; nt < 8; ++nt) o[nt] = {};

    for (int kv0 = 0; kv0 < S_; kv0 += 32) {
        // stage K tile via TDM: fully contiguous 32*128 = 4096 elements
        if (wave == 0) {
            tdm_load_2d((uint32_t)(uintptr_t)&sK[0][0],
                        Kh + ((long)h * S_ + kv0) * D_,
                        /*tile_d0=*/4096, /*tile_d1=*/1,
                        /*tensor_d0=*/0x7fffffffu, /*tensor_d1=*/1,
                        /*stride_d0=*/4096);
        }
        // stage V transposed [d][kv]
        {
            int idx = tid * 16;
            int r = idx >> 7, c = idx & 127;
            const bf16_t* src = V + ((long)h * S_ + kv0 + r) * D_ + c;
            bf16x8 d0 = *(const bf16x8*)(src);
            bf16x8 d1 = *(const bf16x8*)(src + 8);
#pragma unroll
            for (int i = 0; i < 8; ++i) { sVt[c + i][r] = d0[i]; sVt[c + 8 + i][r] = d1[i]; }
        }
        if (wave == 0) tdm_wait0();
        __syncthreads();

        // scores: 16q x 32kv
        floatx8 sc[2];
#pragma unroll
        for (int kt = 0; kt < 2; ++kt) {
            sc[kt] = {};
#pragma unroll
            for (int dk = 0; dk < 4; ++dk) {
                bf16x16 kf = load_frag(&sK[kt * 16][dk * 32], 128);
                sc[kt] = wmma_bf16(qf[dk], kf, sc[kt]);
            }
        }
#pragma unroll
        for (int kt = 0; kt < 2; ++kt)
#pragma unroll
            for (int r = 0; r < 8; ++r) sc[kt][r] *= scale;

        // online softmax; row = r + 8*half, stats reduced over 16 lanes
#pragma unroll
        for (int r = 0; r < 8; ++r) {
            float mx = fmaxf(sc[0][r], sc[1][r]);
            for (int off = 8; off >= 1; off >>= 1)
                mx = fmaxf(mx, __shfl_xor(mx, off));
            float mnew  = fmaxf(m[r], mx);
            float alpha = __expf(m[r] - mnew);
            m[r] = mnew;
            l[r] *= alpha;
#pragma unroll
            for (int nt = 0; nt < 8; ++nt) o[nt][r] *= alpha;

            float p0 = __expf(sc[0][r] - mnew);
            float p1 = __expf(sc[1][r] - mnew);
            float ps = p0 + p1;
            for (int off = 8; off >= 1; off >>= 1)
                ps += __shfl_xor(ps, off);
            l[r] += ps;

            int row = r + 8 * half;
            sP[wave][row][col]      = (bf16_t)p0;
            sP[wave][row][16 + col] = (bf16_t)p1;
        }
        __syncthreads();

        // P(16x32) * V(32x128)
        bf16x16 pf = load_frag(&sP[wave][0][0], 32);
#pragma unroll
        for (int nt = 0; nt < 8; ++nt) {
            bf16x16 vf = load_frag(&sVt[nt * 16][0], 32);
            o[nt] = wmma_bf16(pf, vf, o[nt]);
        }
        __syncthreads();
    }

    // normalize & store bf16 [s][h*D+d]; base pointer + immediate offsets
    bf16_t* ob = O + (long)(qrow0 + 8 * half) * HID_ + h * D_ + col;
#pragma unroll
    for (int r = 0; r < 8; ++r) {
        const float inv = 1.0f / l[r];
        bf16_t* p = ob + (long)r * HID_;
#pragma unroll
        for (int nt = 0; nt < 8; ++nt)
            p[nt * 16] = (bf16_t)(o[nt][r] * inv);
    }
}

// ---------------------------------------------------------------------------
// Host launcher
// ---------------------------------------------------------------------------
extern "C" void kernel_launch(void* const* d_in, const int* in_sizes, int n_in,
                              void* d_out, int out_size, void* d_ws, size_t ws_size,
                              hipStream_t stream) {
    const float* hidden = (const float*)d_in[0];
    const float* fcos   = (const float*)d_in[1];
    const float* fsin   = (const float*)d_in[2];
    const float* w_qkv  = (const float*)d_in[3];
    const float* b_qkv  = (const float*)d_in[4];
    const float* wnq    = (const float*)d_in[5];
    const float* wnk    = (const float*)d_in[6];
    const float* w_out  = (const float*)d_in[7];
    const float* b_out  = (const float*)d_in[8];
    float* out = (float*)d_out;

    char* ws = (char*)d_ws;
    const size_t sz_h    = (size_t)S_ * HID_ * 2;        // 8 MB
    const size_t sz_wqkv = (size_t)HID_ * N3_ * 2;       // 24 MB
    const size_t sz_wout = (size_t)HID_ * HID_ * 2;      // 8 MB
    const size_t sz_qkv  = (size_t)S_ * N3_ * 4;         // 48 MB
    const size_t sz_head = (size_t)H_ * S_ * D_ * 2;     // 8 MB each

    bf16_t* h_bf    = (bf16_t*)(ws);
    bf16_t* wqkv_bf = (bf16_t*)(ws + sz_h);
    bf16_t* wout_bf = (bf16_t*)(ws + sz_h + sz_wqkv);
    float*  qkv_f   = (float*) (ws + sz_h + sz_wqkv + sz_wout);
    bf16_t* q_bf    = (bf16_t*)(ws + sz_h + sz_wqkv + sz_wout + sz_qkv);
    bf16_t* k_bf    = (bf16_t*)(ws + sz_h + sz_wqkv + sz_wout + sz_qkv + sz_head);
    bf16_t* v_bf    = (bf16_t*)(ws + sz_h + sz_wqkv + sz_wout + sz_qkv + 2 * sz_head);
    bf16_t* attn_bf = h_bf;   // hidden_bf16 is dead after the QKV GEMM

    // 1. converts
    long n1 = (long)S_ * HID_;
    long n2 = (long)HID_ * N3_;
    long n3 = (long)HID_ * HID_;
    k_f32_to_bf16<<<(int)(n1 / 1024), 256, 0, stream>>>(hidden, h_bf, n1);
    k_f32_to_bf16<<<(int)(n2 / 1024), 256, 0, stream>>>(w_qkv, wqkv_bf, n2);
    k_f32_to_bf16<<<(int)(n3 / 1024), 256, 0, stream>>>(w_out, wout_bf, n3);

    // 2. QKV GEMM: (2048x2048)*(2048x6144)+b
    k_gemm_bf16<<<dim3(N3_ / 128, S_ / 64), 256, 0, stream>>>(
        h_bf, wqkv_bf, b_qkv, qkv_f, S_, N3_, HID_);

    // 3. RMSNorm + RoPE + head-major transpose
    k_norm_rope<<<S_, 256, 0, stream>>>(qkv_f, fcos, fsin, wnq, wnk, q_bf, k_bf, v_bf);

    // 4. flash attention
    k_attn<<<dim3(S_ / 128, H_), 256, 0, stream>>>(q_bf, k_bf, v_bf, attn_bf);

    // 5. output GEMM: (2048x2048)*(2048x2048)+b
    k_gemm_bf16<<<dim3(HID_ / 128, S_ / 64), 256, 0, stream>>>(
        attn_bf, wout_bf, b_out, out, S_, HID_, HID_);

    (void)in_sizes; (void)n_in; (void)out_size; (void)ws_size;
}
